// TwoWayAttention_29850022708019
// MI455X (gfx1250) — matrix operand order repaired and checked by reading, not compiled, not run on hardware
//
#include <hip/hip_runtime.h>
#include <hip/hip_bf16.h>

typedef __bf16 v16bf __attribute__((ext_vector_type(16)));
typedef float  v8f   __attribute__((ext_vector_type(8)));
typedef int    v4i   __attribute__((ext_vector_type(4)));

constexpr int BATCH = 32;
constexpr int L     = 2048;   // L1 == L2
constexpr int D     = 1024;   // D1 == D2
constexpr int TM = 128, TN = 256, TK = 32;   // workgroup tile; 8 waves of 64x64
constexpr int LDT = 40;       // padded LDS row stride in bf16 elems (80B: 16B-aligned, conflict-free)

// ---------- helpers ----------
__device__ __forceinline__ unsigned short f2bf(float f) {
  return __builtin_bit_cast(unsigned short, (__bf16)f);   // hardware RNE convert
}
__device__ __forceinline__ unsigned pk2(float a, float b) {
#if __has_builtin(__builtin_amdgcn_cvt_pk_bf16_f32)
  return __builtin_bit_cast(unsigned, __builtin_amdgcn_cvt_pk_bf16_f32(a, b));
#else
  return (unsigned)f2bf(a) | ((unsigned)f2bf(b) << 16);
#endif
}
// monotonic float <-> uint mapping so unsigned atomicMax == float max (0 == "smallest")
__device__ __forceinline__ unsigned fenc(float f) {
  unsigned b = __float_as_uint(f);
  return (b & 0x80000000u) ? ~b : (b | 0x80000000u);
}
__device__ __forceinline__ float fdec(unsigned u) {
  unsigned b = (u & 0x80000000u) ? (u ^ 0x80000000u) : ~u;
  return __uint_as_float(b);
}

// CDNA5 async copy: 16B global -> LDS, tracked by ASYNCcnt (no VGPR round-trip).
// LDS operand = low 32 bits of the generic pointer (flat-LDS aperture keeps addr[31:0]).
__device__ __forceinline__ void async_b128(unsigned short* lds_dst, const void* gsrc) {
  asm volatile("global_load_async_to_lds_b128 %0, %1, off"
               :: "v"((unsigned)(uintptr_t)lds_dst),
                  "v"((unsigned long long)(uintptr_t)gsrc)
               : "memory");
}
template<int N>
__device__ __forceinline__ void wait_async() {
  asm volatile("s_wait_asynccnt %0" :: "i"(N) : "memory");
}

// A fragment: 16x32 bf16. lane l = row; hi half: K in {hi*8..hi*8+7, 16+hi*8..16+hi*8+7}
__device__ __forceinline__ v16bf frag_a(const unsigned short* la, int row, int hi) {
  v16bf a;
  ((v4i*)&a)[0] = *(const v4i*)&la[row * LDT + hi * 8];
  ((v4i*)&a)[1] = *(const v4i*)&la[row * LDT + 16 + hi * 8];
  return a;
}
// B fragment: 32x16 bf16. lane l = column; hi half: K in {hi*8..hi*8+7, 16+hi*8..16+hi*8+7}
__device__ __forceinline__ v16bf frag_b(const unsigned short* lb, int col, int hi) {
  v16bf b;
  ((v4i*)&b)[0] = *(const v4i*)&lb[col * LDT + hi * 8];
  ((v4i*)&b)[1] = *(const v4i*)&lb[col * LDT + 16 + hi * 8];
  return b;
}
__device__ __forceinline__ v8f wmma_bf16(v16bf a, v16bf b, v8f c) {
  return __builtin_amdgcn_wmma_f32_16x16x32_bf16(false, a, false, b, (short)0, c, false, false);
}
__device__ __forceinline__ void cvt_store(unsigned short* dst, const float4& f0, const float4& f1,
                                          const float4& f2, const float4& f3) {
  v4i q0, q1;
  q0.x = (int)pk2(f0.x, f0.y); q0.y = (int)pk2(f0.z, f0.w);
  q0.z = (int)pk2(f1.x, f1.y); q0.w = (int)pk2(f1.z, f1.w);
  q1.x = (int)pk2(f2.x, f2.y); q1.y = (int)pk2(f2.z, f2.w);
  q1.z = (int)pk2(f3.x, f3.y); q1.w = (int)pk2(f3.z, f3.w);
  *(v4i*)dst       = q0;
  *(v4i*)(dst + 8) = q1;
}

// ---------- kernel 1: U[k][n] (f32) -> Ut[n][k] (bf16) ----------
__global__ __launch_bounds__(256) void k_transpose_u(const float* __restrict__ U,
                                                     unsigned short* __restrict__ Ut) {
  __shared__ float tile[32][33];
  const int tx = threadIdx.x, ty = threadIdx.y;     // block (32,8)
  const int x0 = blockIdx.x * 32, y0 = blockIdx.y * 32;
  #pragma unroll
  for (int j = 0; j < 32; j += 8)
    tile[ty + j][tx] = U[(size_t)(y0 + ty + j) * D + (x0 + tx)];
  __syncthreads();
  #pragma unroll
  for (int j = 0; j < 32; j += 8)
    Ut[(size_t)(x0 + ty + j) * D + (y0 + tx)] = f2bf(tile[tx][ty + j]);
}

// ---------- kernel 1b: bulk f32 -> bf16 (x2 pre-convert), 8 elems/thread ----------
__global__ __launch_bounds__(256) void k_cvt_bf16(const float* __restrict__ x,
                                                  unsigned short* __restrict__ xb) {
  const size_t i = ((size_t)blockIdx.x * 256 + threadIdx.x) * 8;
  const float4 a = *(const float4*)(x + i);
  const float4 b = *(const float4*)(x + i + 4);
  v4i q;
  q.x = (int)pk2(a.x, a.y); q.y = (int)pk2(a.z, a.w);
  q.z = (int)pk2(b.x, b.y); q.w = (int)pk2(b.z, b.w);
  *(v4i*)(xb + i) = q;
}

// ---------- kernel 2: Y[b] = x1[b] @ U ----------
// A (x1, fp32->bf16): reg-prefetch pipeline into single LDS buffer (128x32).
// B (Ut, bf16 copy): double-buffered LDS via async-to-LDS (256x32).
// 8 waves, 64x64 per wave -> acc[4][4], 16 WMMAs per K-step.
__global__ __launch_bounds__(256) void k_gemm_xu(const float* __restrict__ x1,
                                                 const unsigned short* __restrict__ Ut,
                                                 unsigned short* __restrict__ Y) {
  __shared__ unsigned short la[TM * LDT];
  __shared__ unsigned short lb[2][TN * LDT];
  const int b  = blockIdx.z;
  const int i0 = blockIdx.y * TM;
  const int n0 = blockIdx.x * TN;
  const int tid  = threadIdx.x;
  const int lane = tid & 31, wid = tid >> 5;
  const int wm = wid & 1, wn = wid >> 1;            // 2x4 wave grid -> 64x64 per wave
  const int l  = lane & 15, hi = lane >> 4;
  const int frow = tid >> 1;                        // A fill: 2 threads per row (16 f32 each)
  const int fk   = (tid & 1) * 16;

  const float*          Aa = x1 + ((size_t)b * L + i0 + frow) * D + fk;
  const unsigned short* Bb = Ut + (size_t)(n0 + tid) * D;   // B fill: 1 thread per row (32 bf16)
  unsigned short* lbd0 = &lb[0][tid * LDT];
  unsigned short* lbd1 = &lb[1][tid * LDT];
  unsigned short* lad  = &la[frow * LDT + fk];

  // prologue: async-copy B tile(k=0) into buf0; prefetch A regs for k=0
  #pragma unroll
  for (int o = 0; o < 4; ++o) async_b128(lbd0 + o * 8, Bb + o * 8);
  float4 p0 = *(const float4*)(Aa);
  float4 p1 = *(const float4*)(Aa + 4);
  float4 p2 = *(const float4*)(Aa + 8);
  float4 p3 = *(const float4*)(Aa + 12);

  v8f acc[4][4] = {};
  for (int k0 = 0, it = 0; k0 < D; k0 += TK, ++it) {
    const int  buf  = it & 1;
    const bool more = (k0 + TK) < D;
    __syncthreads();                                // prev compute done: la & lb[buf^1] reusable
    if (more) {                                     // async-fill next B buffer
      unsigned short* d = buf ? lbd0 : lbd1;
      #pragma unroll
      for (int o = 0; o < 4; ++o) async_b128(d + o * 8, Bb + k0 + TK + o * 8);
    }
    cvt_store(lad, p0, p1, p2, p3);                 // stage A for this step
    if (more) {                                     // prefetch A regs for next step
      p0 = *(const float4*)(Aa + k0 + TK);
      p1 = *(const float4*)(Aa + k0 + TK + 4);
      p2 = *(const float4*)(Aa + k0 + TK + 8);
      p3 = *(const float4*)(Aa + k0 + TK + 12);
    }
    if (more) wait_async<4>(); else wait_async<0>();  // current B buffer (older 4) complete
    __syncthreads();
    const unsigned short* lbb = lb[buf];
    v16bf bf[4];
    #pragma unroll
    for (int nf = 0; nf < 4; ++nf) bf[nf] = frag_b(lbb, wn * 64 + nf * 16 + l, hi);
    #pragma unroll
    for (int mf = 0; mf < 4; ++mf) {
      v16bf af = frag_a(la, wm * 64 + mf * 16 + l, hi);
      #pragma unroll
      for (int nf = 0; nf < 4; ++nf) acc[mf][nf] = wmma_bf16(af, bf[nf], acc[mf][nf]);
    }
  }
  #pragma unroll
  for (int mf = 0; mf < 4; ++mf)
    #pragma unroll
    for (int nf = 0; nf < 4; ++nf)
      #pragma unroll
      for (int r = 0; r < 8; ++r) {
        const int row = i0 + wm * 64 + mf * 16 + hi * 8 + r;
        const int col = n0 + wn * 64 + nf * 16 + l;
        Y[((size_t)b * L + row) * D + col] = f2bf(acc[mf][nf][r]);
      }
}

// ---------- kernel 3: tile of M = Y @ x2^T; fused row/col max ----------
// Both operands bf16 -> pure async-to-LDS fills, both double-buffered. No ds_stores in loop.
__global__ __launch_bounds__(256) void k_gemm_max(const unsigned short* __restrict__ Y,
                                                  const unsigned short* __restrict__ x2b,
                                                  unsigned* __restrict__ rowmaxu,
                                                  unsigned* __restrict__ colmaxu) {
  __shared__ unsigned short la[2][TM * LDT];
  __shared__ unsigned short lb[2][TN * LDT];
  __shared__ unsigned srow[TM];
  __shared__ unsigned scol[TN];
  const int b  = blockIdx.z;
  const int i0 = blockIdx.y * TM;
  const int j0 = blockIdx.x * TN;
  const int tid  = threadIdx.x;
  const int lane = tid & 31, wid = tid >> 5;
  const int wm = wid & 1, wn = wid >> 1;
  const int l  = lane & 15, hi = lane >> 4;
  const int frow = tid >> 1;
  const int fk   = (tid & 1) * 16;

  const unsigned short* Aa = Y   + ((size_t)b * L + i0 + frow) * D + fk;  // 2 threads/row
  const unsigned short* Bb = x2b + ((size_t)b * L + j0 + tid) * D;        // 1 thread/row
  unsigned short* lad0 = &la[0][frow * LDT + fk];
  unsigned short* lad1 = &la[1][frow * LDT + fk];
  unsigned short* lbd0 = &lb[0][tid * LDT];
  unsigned short* lbd1 = &lb[1][tid * LDT];

  srow[tid & (TM - 1)] = 0u;   // tid 0..255 covers srow[128] twice (benign) ...
  scol[tid] = 0u;              // ... and scol[256] once

  // prologue: fill buf0 for both operands
  async_b128(lad0,     Aa);
  async_b128(lad0 + 8, Aa + 8);
  #pragma unroll
  for (int o = 0; o < 4; ++o) async_b128(lbd0 + o * 8, Bb + o * 8);

  v8f acc[4][4] = {};
  for (int k0 = 0, it = 0; k0 < D; k0 += TK, ++it) {
    const int  buf  = it & 1;
    const bool more = (k0 + TK) < D;
    __syncthreads();                                 // prev compute done: buf^1 reusable
    if (more) {
      unsigned short* da = buf ? lad0 : lad1;
      unsigned short* db = buf ? lbd0 : lbd1;
      async_b128(da,     Aa + k0 + TK);
      async_b128(da + 8, Aa + k0 + TK + 8);
      #pragma unroll
      for (int o = 0; o < 4; ++o) async_b128(db + o * 8, Bb + k0 + TK + o * 8);
    }
    if (more) wait_async<6>(); else wait_async<0>(); // current buffers (older 6) complete
    __syncthreads();                                 // all waves' fills of buf visible
    const unsigned short* laa = la[buf];
    const unsigned short* lbb = lb[buf];
    v16bf bf[4];
    #pragma unroll
    for (int nf = 0; nf < 4; ++nf) bf[nf] = frag_b(lbb, wn * 64 + nf * 16 + l, hi);
    #pragma unroll
    for (int mf = 0; mf < 4; ++mf) {
      v16bf af = frag_a(laa, wm * 64 + mf * 16 + l, hi);
      #pragma unroll
      for (int nf = 0; nf < 4; ++nf) acc[mf][nf] = wmma_bf16(af, bf[nf], acc[mf][nf]);
    }
  }
  // ---- fused reductions: C/D layout: M = 8*hi + r, N = l ----
  #pragma unroll
  for (int mf = 0; mf < 4; ++mf)
    #pragma unroll
    for (int r = 0; r < 8; ++r) {
      float v = fmaxf(fmaxf(acc[mf][0][r], acc[mf][1][r]),
                      fmaxf(acc[mf][2][r], acc[mf][3][r]));
      atomicMax(&srow[wm * 64 + mf * 16 + hi * 8 + r], fenc(v));
    }
  #pragma unroll
  for (int nf = 0; nf < 4; ++nf) {
    float v = -3.4e38f;
    #pragma unroll
    for (int mf = 0; mf < 4; ++mf)
      #pragma unroll
      for (int r = 0; r < 8; ++r) v = fmaxf(v, acc[mf][nf][r]);
    atomicMax(&scol[wn * 64 + nf * 16 + l], fenc(v));
  }
  __syncthreads();
  if (tid < TM) atomicMax(&rowmaxu[(size_t)b * L + i0 + tid], srow[tid]);
  atomicMax(&colmaxu[(size_t)b * L + j0 + tid], scol[tid]);
}

// ---------- kernel 4: softmax over decoded maxes, one block per batch ----------
__global__ __launch_bounds__(256) void k_softmax(const unsigned* __restrict__ mu,
                                                 float* __restrict__ w) {
  __shared__ float red[256];
  const int b = blockIdx.x, tid = threadIdx.x;
  const unsigned* m = mu + (size_t)b * L;
  float mx = -3.4e38f;
  for (int i = tid; i < L; i += 256) mx = fmaxf(mx, fdec(m[i]));
  red[tid] = mx; __syncthreads();
  for (int s = 128; s > 0; s >>= 1) { if (tid < s) red[tid] = fmaxf(red[tid], red[tid + s]); __syncthreads(); }
  mx = red[0]; __syncthreads();
  float sum = 0.f;
  for (int i = tid; i < L; i += 256) sum += __expf(fdec(m[i]) - mx);
  red[tid] = sum; __syncthreads();
  for (int s = 128; s > 0; s >>= 1) { if (tid < s) red[tid] += red[tid + s]; __syncthreads(); }
  const float inv = 1.0f / red[0];
  for (int i = tid; i < L; i += 256) w[(size_t)b * L + i] = __expf(fdec(m[i]) - mx) * inv;
}

// ---------- kernel 5: out[b][d] = sum_i w[b][i] * x[b][i][d] ----------
__global__ __launch_bounds__(256) void k_wsum(const float* __restrict__ w,
                                              const float* __restrict__ x,
                                              float* __restrict__ out) {
  const int b = blockIdx.y;
  const int d = blockIdx.x * 256 + threadIdx.x;
  const float* xb = x + (size_t)b * L * D + d;
  const float* wb = w + (size_t)b * L;
  float acc = 0.f;
  for (int i = 0; i < L; ++i) acc = fmaf(wb[i], xb[(size_t)i * D], acc);
  out[(size_t)b * D + d] = acc;
}

extern "C" void kernel_launch(void* const* d_in, const int* in_sizes, int n_in,
                              void* d_out, int out_size, void* d_ws, size_t ws_size,
                              hipStream_t stream) {
  (void)in_sizes; (void)n_in; (void)out_size; (void)ws_size;
  const float* x1 = (const float*)d_in[0];
  const float* x2 = (const float*)d_in[1];
  const float* U  = (const float*)d_in[2];
  float* out = (float*)d_out;

  char* ws = (char*)d_ws;
  size_t off = 0;
  auto alloc = [&](size_t bytes) -> void* {
    void* p = ws + off; off = (off + bytes + 255) & ~(size_t)255; return p;
  };
  unsigned short* Ut      = (unsigned short*)alloc((size_t)D * D * 2);          //   2 MB
  unsigned short* Ybf     = (unsigned short*)alloc((size_t)BATCH * L * D * 2);  // 128 MB
  unsigned short* x2bf    = (unsigned short*)alloc((size_t)BATCH * L * D * 2);  // 128 MB
  unsigned*       rowmaxu = (unsigned*)alloc((size_t)BATCH * L * 4);
  unsigned*       colmaxu = (unsigned*)alloc((size_t)BATCH * L * 4);
  float*          rows    = (float*)alloc((size_t)BATCH * L * 4);
  float*          cols    = (float*)alloc((size_t)BATCH * L * 4);

  hipMemsetAsync(rowmaxu, 0, (size_t)BATCH * L * 4, stream);
  hipMemsetAsync(colmaxu, 0, (size_t)BATCH * L * 4, stream);

  k_transpose_u<<<dim3(D / 32, D / 32), dim3(32, 8), 0, stream>>>(U, Ut);
  k_cvt_bf16<<<dim3((unsigned)((size_t)BATCH * L * D / (256 * 8))), 256, 0, stream>>>(x2, x2bf);
  k_gemm_xu <<<dim3(D / TN, L / TM, BATCH), 256, 0, stream>>>(x1, Ut, Ybf);
  k_gemm_max<<<dim3(L / TN, L / TM, BATCH), 256, 0, stream>>>(Ybf, x2bf, rowmaxu, colmaxu);
  k_softmax<<<BATCH, 256, 0, stream>>>(rowmaxu, rows);
  k_softmax<<<BATCH, 256, 0, stream>>>(colmaxu, cols);
  k_wsum<<<dim3(D / 256, BATCH), 256, 0, stream>>>(rows, x1, out);
  k_wsum<<<dim3(D / 256, BATCH), 256, 0, stream>>>(cols, x2, out + (size_t)BATCH * D);
}
